// LocalAttention_37383395344552
// MI455X (gfx1250) — compile-verified, matrix operand order checked by reading, not run on hardware
//
#include <hip/hip_runtime.h>
#include <hip/hip_bf16.h>

typedef __attribute__((ext_vector_type(16))) __bf16 v16bf;
typedef __attribute__((ext_vector_type(8)))  __bf16 v8bf;
typedef __attribute__((ext_vector_type(4)))  __bf16 v4bf;
typedef __attribute__((ext_vector_type(8)))  float  v8f;

// ---------------------------------------------------------------------------
// fp32 -> bf16 conversion (x and weight matrices), 4 elements / thread
// ---------------------------------------------------------------------------
__global__ __launch_bounds__(256)
void cvt_f32_bf16_kernel(const float* __restrict__ in, __bf16* __restrict__ out, int n) {
    int i = (blockIdx.x * blockDim.x + threadIdx.x) * 4;
    if (i < n) {
        float4 f = *reinterpret_cast<const float4*>(in + i);
        v4bf o;
        o.x = (__bf16)f.x; o.y = (__bf16)f.y; o.z = (__bf16)f.z; o.w = (__bf16)f.w;
        *reinterpret_cast<v4bf*>(out + i) = o;
    }
}

// ---------------------------------------------------------------------------
// Fragment loader: 16x32 bf16 tile, documented CDNA5 16-bit A/B layout.
// Per lane: 8 contiguous bf16 at p, 8 contiguous bf16 at p+16 (two b128 loads).
// ---------------------------------------------------------------------------
__device__ __forceinline__ v16bf load_frag(const __bf16* p) {
    v16bf f;
    *reinterpret_cast<v8bf*>(&f)       = *reinterpret_cast<const v8bf*>(p);
    *(reinterpret_cast<v8bf*>(&f) + 1) = *reinterpret_cast<const v8bf*>(p + 16);
    return f;
}

__device__ __forceinline__ v8f wmma_bf16(v16bf a, v16bf b, v8f c) {
    return __builtin_amdgcn_wmma_f32_16x16x32_bf16(
        false, a, false, b, (short)0, c, false, false);
}

// ---------------------------------------------------------------------------
// GEMM: Y[M,N] = X[M,K] * Wt[N,K]^T + bias  (torch Linear: y = x @ W.T + b)
// One wave computes a 32x64 output tile (2 M-tiles x 4 N-tiles = 8 WMMAs per
// K-step of 32). Fragments are double-buffered so the next K-step's loads are
// in flight while the current step's WMMAs execute (partial s_wait_loadcnt
// instead of a full drain).
// ---------------------------------------------------------------------------
template<bool BF16OUT>
__global__ __launch_bounds__(128)
void gemm_xwt_kernel(const __bf16* __restrict__ X, const __bf16* __restrict__ Wt,
                     const float* __restrict__ bias, void* __restrict__ Y,
                     int M, int N, int K)
{
    const int lane   = threadIdx.x & 31;
    const int wave   = blockIdx.x * (blockDim.x >> 5) + (threadIdx.x >> 5);
    const int nTiles = N >> 6;                   // groups of 64 columns
    const int row0   = (wave / nTiles) << 5;     // 32 rows per wave
    const int col0   = (wave % nTiles) << 6;     // 64 cols per wave
    const int half   = lane >> 4;                // 0: lanes 0-15, 1: lanes 16-31
    const int l16    = lane & 15;
    const int kSel   = half << 3;                // K sub-offset per ISA layout

    const __bf16* aRow0 = X  + (size_t)(row0 + l16) * K + kSel;
    const __bf16* aRow1 = X  + (size_t)(row0 + 16 + l16) * K + kSel;
    const __bf16* bRow  = Wt + (size_t)(col0 + l16) * K + kSel;

    v8f acc[8] = {};

    // -------- prologue: load K-step 0 fragments --------
    v16bf a0 = load_frag(aRow0);
    v16bf a1 = load_frag(aRow1);
    v16bf b0 = load_frag(bRow);
    v16bf b1 = load_frag(bRow + (size_t)16 * K);
    v16bf b2 = load_frag(bRow + (size_t)32 * K);
    v16bf b3 = load_frag(bRow + (size_t)48 * K);

    // -------- pipelined main loop --------
    for (int kk = 32; kk < K; kk += 32) {
        v16bf a0n = load_frag(aRow0 + kk);
        v16bf a1n = load_frag(aRow1 + kk);
        v16bf b0n = load_frag(bRow + kk);
        v16bf b1n = load_frag(bRow + (size_t)16 * K + kk);
        v16bf b2n = load_frag(bRow + (size_t)32 * K + kk);
        v16bf b3n = load_frag(bRow + (size_t)48 * K + kk);

        acc[0] = wmma_bf16(a0, b0, acc[0]);
        acc[1] = wmma_bf16(a0, b1, acc[1]);
        acc[2] = wmma_bf16(a0, b2, acc[2]);
        acc[3] = wmma_bf16(a0, b3, acc[3]);
        acc[4] = wmma_bf16(a1, b0, acc[4]);
        acc[5] = wmma_bf16(a1, b1, acc[5]);
        acc[6] = wmma_bf16(a1, b2, acc[6]);
        acc[7] = wmma_bf16(a1, b3, acc[7]);

        a0 = a0n; a1 = a1n;
        b0 = b0n; b1 = b1n; b2 = b2n; b3 = b3n;
    }

    // -------- final K-step --------
    acc[0] = wmma_bf16(a0, b0, acc[0]);
    acc[1] = wmma_bf16(a0, b1, acc[1]);
    acc[2] = wmma_bf16(a0, b2, acc[2]);
    acc[3] = wmma_bf16(a0, b3, acc[3]);
    acc[4] = wmma_bf16(a1, b0, acc[4]);
    acc[5] = wmma_bf16(a1, b1, acc[5]);
    acc[6] = wmma_bf16(a1, b2, acc[6]);
    acc[7] = wmma_bf16(a1, b3, acc[7]);

    // -------- epilogue: C/D layout -> lane = column, VGPR r = row (+8 upper) --------
#pragma unroll
    for (int mt = 0; mt < 2; ++mt) {
#pragma unroll
        for (int t = 0; t < 4; ++t) {
            const int col = col0 + t * 16 + l16;
            const float bv = bias[col];
            const v8f& c = acc[mt * 4 + t];
#pragma unroll
            for (int r = 0; r < 8; ++r) {
                const int row = row0 + mt * 16 + r + (half << 3);
                const float v = c[r] + bv;
                if (BF16OUT) reinterpret_cast<__bf16*>(Y)[(size_t)row * N + col] = (__bf16)v;
                else         reinterpret_cast<float*>(Y)[(size_t)row * N + col] = v;
            }
        }
    }
}

// ---------------------------------------------------------------------------
// Local windowed attention (window=5, hd=64). One wave32 per (b, i, h).
// Lane owns head dims {lane, lane+32}; scores via shfl_xor tree reduction.
// ---------------------------------------------------------------------------
__global__ __launch_bounds__(256)
void local_attn_kernel(const __bf16* __restrict__ Q, const __bf16* __restrict__ Kc,
                       const __bf16* __restrict__ V, __bf16* __restrict__ O,
                       int S, int H, int D)
{
    const int lane = threadIdx.x & 31;
    const int w    = blockIdx.x * (blockDim.x >> 5) + (threadIdx.x >> 5);
    const int h    = w % H;
    const int bs   = w / H;          // b*S + i
    const int i    = bs % S;

    const size_t rowBase = (size_t)bs * D + h * 64;
    const float q0 = (float)Q[rowBase + lane];
    const float q1 = (float)Q[rowBase + lane + 32];

    const int jlo = (i - 2 < 0) ? 0 : i - 2;
    const int jhi = (i + 2 >= S) ? S - 1 : i + 2;
    const int cnt = jhi - jlo + 1;

    float p[5];
    float mx = -1e30f;
    for (int n = 0; n < cnt; ++n) {
        const size_t kRow = (size_t)(bs - i + jlo + n) * D + h * 64;
        float partial = q0 * (float)Kc[kRow + lane] + q1 * (float)Kc[kRow + lane + 32];
#pragma unroll
        for (int off = 16; off > 0; off >>= 1)
            partial += __shfl_xor(partial, off, 32);
        const float s = partial * 0.125f;   // 1/sqrt(64)
        p[n] = s;
        mx = fmaxf(mx, s);
    }
    float sum = 0.f;
    for (int n = 0; n < cnt; ++n) { p[n] = __expf(p[n] - mx); sum += p[n]; }
    const float inv = 1.f / sum;

    float o0 = 0.f, o1 = 0.f;
    for (int n = 0; n < cnt; ++n) {
        const size_t vRow = (size_t)(bs - i + jlo + n) * D + h * 64;
        const float pw = p[n] * inv;
        o0 += pw * (float)V[vRow + lane];
        o1 += pw * (float)V[vRow + lane + 32];
    }
    O[rowBase + lane]      = (__bf16)o0;
    O[rowBase + lane + 32] = (__bf16)o1;
}

// ---------------------------------------------------------------------------
// Host-side launch
// ---------------------------------------------------------------------------
extern "C" void kernel_launch(void* const* d_in, const int* in_sizes, int n_in,
                              void* d_out, int out_size, void* d_ws, size_t ws_size,
                              hipStream_t stream) {
    const float* x  = (const float*)d_in[0];
    const float* Wq = (const float*)d_in[1];
    const float* bq = (const float*)d_in[2];
    const float* Wk = (const float*)d_in[3];
    const float* bk = (const float*)d_in[4];
    const float* Wv = (const float*)d_in[5];
    const float* bv = (const float*)d_in[6];
    const float* Wo = (const float*)d_in[7];
    const float* bo = (const float*)d_in[8];
    float* out = (float*)d_out;

    constexpr int B = 2, S = 2048, H = 16, D = 1024;
    constexpr int M = B * S;          // 4096 rows
    constexpr int K = D, N = D;
    constexpr size_t XB_ELEMS = (size_t)M * D;      // 4,194,304
    constexpr size_t WB_ELEMS = (size_t)D * D;      // 1,048,576

    char* ws = (char*)d_ws;
    __bf16* xb  = (__bf16*)(ws);
    __bf16* wqb = (__bf16*)(ws + 8388608);
    __bf16* wkb = (__bf16*)(ws + 10485760);
    __bf16* wvb = (__bf16*)(ws + 12582912);
    __bf16* wob = (__bf16*)(ws + 14680064);
    __bf16* Qb  = (__bf16*)(ws + 16777216);
    __bf16* Kb  = (__bf16*)(ws + 25165824);
    __bf16* Vb  = (__bf16*)(ws + 33554432);
    __bf16* Ob  = (__bf16*)(ws + 41943040);

    // 1) convert inputs to bf16
    {
        int nX = (int)XB_ELEMS, nW = (int)WB_ELEMS;
        cvt_f32_bf16_kernel<<<(nX / 4 + 255) / 256, 256, 0, stream>>>(x,  xb,  nX);
        cvt_f32_bf16_kernel<<<(nW / 4 + 255) / 256, 256, 0, stream>>>(Wq, wqb, nW);
        cvt_f32_bf16_kernel<<<(nW / 4 + 255) / 256, 256, 0, stream>>>(Wk, wkb, nW);
        cvt_f32_bf16_kernel<<<(nW / 4 + 255) / 256, 256, 0, stream>>>(Wv, wvb, nW);
        cvt_f32_bf16_kernel<<<(nW / 4 + 255) / 256, 256, 0, stream>>>(Wo, wob, nW);
    }

    // 2) Q/K/V projections via WMMA (bf16 in, f32 acc, bf16 out)
    const int gemmWaves  = (M / 32) * (N / 64);   // 2048 waves, 32x64 tile each
    const int gemmBlocks = gemmWaves / 4;         // 512 blocks of 4 waves
    gemm_xwt_kernel<true><<<gemmBlocks, 128, 0, stream>>>(xb, wqb, bq, Qb, M, N, K);
    gemm_xwt_kernel<true><<<gemmBlocks, 128, 0, stream>>>(xb, wkb, bk, Kb, M, N, K);
    gemm_xwt_kernel<true><<<gemmBlocks, 128, 0, stream>>>(xb, wvb, bv, Vb, M, N, K);

    // 3) local windowed attention (wave per (b,i,h))
    const int attnWaves  = M * H;                 // 65536
    const int attnBlocks = attnWaves / 8;         // 8192 blocks of 8 waves
    local_attn_kernel<<<attnBlocks, 256, 0, stream>>>(Qb, Kb, Vb, Ob, S, H, D);

    // 4) output projection, fp32 result (+ bias) straight into d_out
    gemm_xwt_kernel<false><<<gemmBlocks, 128, 0, stream>>>(Ob, wob, bo, out, M, N, K);

    (void)in_sizes; (void)n_in; (void)out_size; (void)ws_size;
}
